// ResNet_43387759624234
// MI455X (gfx1250) — matrix operand order repaired and checked by reading, not compiled, not run on hardware
//
#include <hip/hip_runtime.h>

// ---------------------------------------------------------------------------
// Sparse submanifold conv ResNet (C=32, K=9 taps, NB residual blocks) on
// MI455X / gfx1250. Each conv is a gather-GEMM done with
// v_wmma_f32_16x16x32_bf16: per 16-site tile, 9 taps x 2 output-channel
// tiles of 16x16x32 WMMA, f32 accumulate. Weights pre-packed into the
// documented wave32 B-matrix layout and staged in LDS.
// ---------------------------------------------------------------------------

typedef __bf16 bf16_t;
typedef __attribute__((ext_vector_type(16))) __bf16 v16bf;
typedef __attribute__((ext_vector_type(8)))  __bf16 v8bf;
typedef __attribute__((ext_vector_type(8)))  float  v8f;

#define C_CH 32
#define KTAP 9
#define WPC  (KTAP * 2 * 32 * 16)   // packed bf16 weight elems per conv = 9216
#define STAT_BLOCKS 512
#define STAT_THREADS 256

// ---- init: f32 feats -> padded bf16 buffer; zero the padding rows ---------
__global__ void init_feats(const float* __restrict__ feats,
                           bf16_t* __restrict__ xb, bf16_t* __restrict__ tb,
                           int n_elems, int tot_elems) {
  int idx = blockIdx.x * blockDim.x + threadIdx.x;
  if (idx >= tot_elems) return;
  if (idx < n_elems) {
    xb[idx] = (bf16_t)feats[idx];
  } else {            // zero pad row (gather target for missing neighbors)
    xb[idx] = (bf16_t)0.0f;
    tb[idx] = (bf16_t)0.0f;
  }
}

// ---- pack f32 weights [NB,K,C,C] into wave32 WMMA B-layout (bf16) ---------
// packed index e = (((conv*9 + k)*2 + ntile)*32 + lane)*16 + elem
// B[K=cin][N=cout]: lane&15 = cout column, lane>>4 selects K half,
// elem i -> cin = (lane>>4)*8 + (i&7) + (i>>3)*16 ; cout = ntile*16 + (lane&15)
__global__ void pack_weights(const float* __restrict__ w1,
                             const float* __restrict__ w2,
                             bf16_t* __restrict__ wp, int nb_blocks) {
  int e = blockIdx.x * blockDim.x + threadIdx.x;
  int total = nb_blocks * 2 * WPC;
  if (e >= total) return;
  int i  = e & 15;
  int l  = (e >> 4) & 31;
  int t  = (e >> 9) & 1;
  int k  = (e >> 10) % KTAP;
  int j  = e / WPC;            // conv index: 0..2*NB-1
  int blk = j >> 1;
  const float* w = (j & 1) ? w2 : w1;
  int cin  = ((l >> 4) << 3) + (i & 7) + ((i >> 3) << 4);
  int cout = (t << 4) + (l & 15);
  wp[e] = (bf16_t)w[((size_t)(blk * KTAP + k) * C_CH + cin) * C_CH + cout];
}

// ---- conv: gather + WMMA GEMM ---------------------------------------------
// One wave per 16-site tile; both 16-wide output-channel tiles accumulated.
__global__ __launch_bounds__(256) void conv_wmma(
    const bf16_t* __restrict__ xpad,   // [(N+1) x 32] bf16, row N zeroed
    const int*    __restrict__ nbr,    // [N x 9] neighbor index or N (pad)
    const bf16_t* __restrict__ wconv,  // packed weights for this conv (WPC)
    float*        __restrict__ out,    // [N x 32] f32
    int nsites) {
  __shared__ __align__(16) bf16_t sw[WPC];   // 18 KB of LDS per block
  {
    const uint4* src = (const uint4*)wconv;
    uint4*       dst = (uint4*)sw;
    for (int u = threadIdx.x; u < WPC / 8; u += blockDim.x) dst[u] = src[u];
  }
  __syncthreads();

  int lane = threadIdx.x & 31;
  int wave = threadIdx.x >> 5;
  int tile = blockIdx.x * 8 + wave;
  int ntiles = nsites >> 4;          // nsites is a multiple of 16
  if (tile >= ntiles) return;        // wave-uniform exit (EXEC all-ones for WMMA)

  int row = lane & 15;               // site row within tile (A-matrix M)
  int hf  = lane >> 4;               // K-half select per A layout
  int site = tile * 16 + row;
  const int* nrow = nbr + (size_t)site * KTAP;

  v8f c0 = {};                       // output cols 0..15
  v8f c1 = {};                       // output cols 16..31

#pragma unroll
  for (int k = 0; k < KTAP; ++k) {
    int nb = nrow[k];                // == nsites for missing neighbor (zero row)
    const bf16_t* ap = xpad + (size_t)nb * C_CH + hf * 8;
    v8bf alo = *(const v8bf*)ap;          // channels hf*8 .. hf*8+7
    v8bf ahi = *(const v8bf*)(ap + 16);   // channels hf*8+16 .. hf*8+23
    v16bf a;
#pragma unroll
    for (int i = 0; i < 8; ++i) { a[i] = alo[i]; a[i + 8] = ahi[i]; }

    const bf16_t* b0p = sw + ((size_t)(k * 2 + 0) * 32 + lane) * 16;
    const bf16_t* b1p = sw + ((size_t)(k * 2 + 1) * 32 + lane) * 16;
    v8bf q0 = *(const v8bf*)b0p;
    v8bf q1 = *(const v8bf*)(b0p + 8);
    v8bf q2 = *(const v8bf*)b1p;
    v8bf q3 = *(const v8bf*)(b1p + 8);
    v16bf b0, b1;
#pragma unroll
    for (int i = 0; i < 8; ++i) {
      b0[i] = q0[i]; b0[i + 8] = q1[i];
      b1[i] = q2[i]; b1[i + 8] = q3[i];
    }

    c0 = __builtin_amdgcn_wmma_f32_16x16x32_bf16(false, a, false, b0,
                                                 (short)0, c0, false, false);
    c1 = __builtin_amdgcn_wmma_f32_16x16x32_bf16(false, a, false, b1,
                                                 (short)0, c1, false, false);
  }

  // C/D layout: VGPR r -> row hf*8 + r, lanes 0..15 -> columns 0..15
  float* orow = out + ((size_t)tile * 16 + hf * 8) * C_CH;
#pragma unroll
  for (int r = 0; r < 8; ++r) {
    orow[(size_t)r * C_CH + (lane & 15)]      = c0[r];
    orow[(size_t)r * C_CH + 16 + (lane & 15)] = c1[r];
  }
}

// ---- deterministic per-channel sum / sumsq partials -----------------------
__global__ __launch_bounds__(STAT_THREADS) void stats_partial(
    const float* __restrict__ t, float* __restrict__ partials, int n_elems) {
  __shared__ float ssum[STAT_THREADS];
  __shared__ float ssq[STAT_THREADS];
  float s = 0.0f, q = 0.0f;
  int stride = gridDim.x * blockDim.x;           // multiple of 32
  for (int idx = blockIdx.x * blockDim.x + threadIdx.x; idx < n_elems;
       idx += stride) {
    float v = t[idx];                            // channel = idx & 31 = tid & 31
    s += v; q += v * v;
  }
  ssum[threadIdx.x] = s;
  ssq[threadIdx.x]  = q;
  __syncthreads();
  if (threadIdx.x < 32) {
    float as = 0.0f, aq = 0.0f;
    for (int j = threadIdx.x; j < STAT_THREADS; j += 32) {
      as += ssum[j]; aq += ssq[j];
    }
    partials[(size_t)blockIdx.x * 64 + threadIdx.x]      = as;
    partials[(size_t)blockIdx.x * 64 + 32 + threadIdx.x] = aq;
  }
}

// ---- fold partials -> BN scale/shift --------------------------------------
__global__ void bn_finalize(const float* __restrict__ partials,
                            const float* __restrict__ gamma,
                            const float* __restrict__ beta,
                            float* __restrict__ ss, int nsites, int nblocks) {
  int c = threadIdx.x;
  if (c >= 32) return;
  float s = 0.0f, q = 0.0f;
  for (int b = 0; b < nblocks; ++b) {
    s += partials[(size_t)b * 64 + c];
    q += partials[(size_t)b * 64 + 32 + c];
  }
  float inv_n = 1.0f / (float)nsites;
  float mean = s * inv_n;
  float var  = q * inv_n - mean * mean;
  float scale = gamma[c] * rsqrtf(var + 1e-4f);
  ss[c]      = scale;
  ss[32 + c] = beta[c] - mean * scale;
}

// ---- BN + ReLU -> bf16 intermediate (conv2 input) -------------------------
__global__ void bn_relu_to_bf16(const float* __restrict__ t,
                                const float* __restrict__ ss,
                                bf16_t* __restrict__ outb, int n_elems) {
  int idx = blockIdx.x * blockDim.x + threadIdx.x;
  if (idx >= n_elems) return;
  int c = idx & 31;
  float v = fmaf(t[idx], ss[c], ss[32 + c]);
  v = v > 0.0f ? v : 0.0f;
  outb[idx] = (bf16_t)v;
}

// ---- BN + residual + ReLU -> f32 out (and bf16 copy for next gather) ------
__global__ void bn_add_relu(const float* __restrict__ t,
                            const float* __restrict__ ss,
                            const float* __restrict__ res,
                            float* __restrict__ outf,
                            bf16_t* __restrict__ outb, int write_b,
                            int n_elems) {
  int idx = blockIdx.x * blockDim.x + threadIdx.x;
  if (idx >= n_elems) return;
  int c = idx & 31;
  float v = fmaf(t[idx], ss[c], ss[32 + c]) + res[idx];
  v = v > 0.0f ? v : 0.0f;
  outf[idx] = v;
  if (write_b) outb[idx] = (bf16_t)v;
}

// ---------------------------------------------------------------------------
extern "C" void kernel_launch(void* const* d_in, const int* in_sizes, int n_in,
                              void* d_out, int out_size, void* d_ws,
                              size_t ws_size, hipStream_t stream) {
  const float* feats = (const float*)d_in[0];
  const int*   nbr   = (const int*)d_in[1];
  const float* w1    = (const float*)d_in[2];
  const float* w2    = (const float*)d_in[3];
  const float* g1    = (const float*)d_in[4];
  const float* b1    = (const float*)d_in[5];
  const float* g2    = (const float*)d_in[6];
  const float* b2    = (const float*)d_in[7];
  float* outp = (float*)d_out;

  int n_elems   = in_sizes[0];                        // N * 32
  int nsites    = n_elems / C_CH;                     // N (multiple of 16)
  int nb_blocks = in_sizes[2] / (KTAP * C_CH * C_CH); // NB

  // workspace carve-up (256B aligned slabs)
  char* ws = (char*)d_ws;
  size_t off = 0;
  bf16_t* wp = (bf16_t*)(ws + off);
  off += (size_t)nb_blocks * 2 * WPC * sizeof(bf16_t); off = (off + 255) & ~(size_t)255;
  float* ss = (float*)(ws + off);
  off += 64 * sizeof(float);                           off = (off + 255) & ~(size_t)255;
  float* partials = (float*)(ws + off);
  off += (size_t)STAT_BLOCKS * 64 * sizeof(float);     off = (off + 255) & ~(size_t)255;
  bf16_t* xb = (bf16_t*)(ws + off);                    // padded bf16 features
  off += (size_t)(nsites + 1) * C_CH * sizeof(bf16_t); off = (off + 255) & ~(size_t)255;
  bf16_t* tb = (bf16_t*)(ws + off);                    // padded bf16 intermediate
  off += (size_t)(nsites + 1) * C_CH * sizeof(bf16_t); off = (off + 255) & ~(size_t)255;
  float* t = (float*)(ws + off);                       // f32 conv output
  off += (size_t)nsites * C_CH * sizeof(float);

  int tot_elems = (nsites + 1) * C_CH;
  int initGrid  = (tot_elems + 255) / 256;
  int packTotal = nb_blocks * 2 * WPC;
  int packGrid  = (packTotal + 255) / 256;
  int ntiles    = nsites / 16;
  int convGrid  = (ntiles + 7) / 8;                    // 8 waves per block
  int eltGrid   = (n_elems + 255) / 256;

  init_feats<<<initGrid, 256, 0, stream>>>(feats, xb, tb, n_elems, tot_elems);
  pack_weights<<<packGrid, 256, 0, stream>>>(w1, w2, wp, nb_blocks);

  for (int blk = 0; blk < nb_blocks; ++blk) {
    const bf16_t* wc1 = wp + (size_t)(2 * blk) * WPC;
    const bf16_t* wc2 = wp + (size_t)(2 * blk + 1) * WPC;

    // conv1 + BN + ReLU
    conv_wmma<<<convGrid, 256, 0, stream>>>(xb, nbr, wc1, t, nsites);
    stats_partial<<<STAT_BLOCKS, STAT_THREADS, 0, stream>>>(t, partials, n_elems);
    bn_finalize<<<1, 32, 0, stream>>>(partials, g1 + blk * C_CH, b1 + blk * C_CH,
                                      ss, nsites, STAT_BLOCKS);
    bn_relu_to_bf16<<<eltGrid, 256, 0, stream>>>(t, ss, tb, n_elems);

    // conv2 + BN + residual + ReLU
    conv_wmma<<<convGrid, 256, 0, stream>>>(tb, nbr, wc2, t, nsites);
    stats_partial<<<STAT_BLOCKS, STAT_THREADS, 0, stream>>>(t, partials, n_elems);
    bn_finalize<<<1, 32, 0, stream>>>(partials, g2 + blk * C_CH, b2 + blk * C_CH,
                                      ss, nsites, STAT_BLOCKS);
    const float* res = (blk == 0) ? feats : outp;      // residual input
    bn_add_relu<<<eltGrid, 256, 0, stream>>>(t, ss, res, outp, xb,
                                             (blk + 1 < nb_blocks) ? 1 : 0,
                                             n_elems);
  }
}
